// Critic_23862838297042
// MI455X (gfx1250) — matrix operand order repaired and checked by reading, not compiled, not run on hardware
//
#include <hip/hip_runtime.h>
#include <hip/hip_bf16.h>

typedef __attribute__((ext_vector_type(16))) _Float16 v16h;
typedef __attribute__((ext_vector_type(8)))  _Float16 v8h;
typedef __attribute__((ext_vector_type(8)))  float    v8f;

#define BATCH 1024
#define EDIM  200
#define OC    32
#define FW    9
#define WP    192      // EDIM - FW + 1
#define FCIN  6144     // OC * WP
#define NENT  50000
#define KPAD  224      // EDIM padded to multiple of 32
#define NT_FC 13       // ceil(200/16)
#define KS_FC 192      // 6144/32
#define MT    64       // 1024/16
#define NT_LG 3125     // 50000/16
#define KS_LG 7        // 224/32
#define EPSV  1e-5f

// ---------------------------------------------------------------------------
// Stage 1: gather + e1 = [es|ec]@W1 + bn0, k = r@fc1 (bn1 folded), per-sample
// conv -> flat (f16).  One block per batch row.
// ---------------------------------------------------------------------------
__global__ __launch_bounds__(288)
void k_front(const int* __restrict__ e_s, const int* __restrict__ q_s,
             const int* __restrict__ e_c,
             const float* __restrict__ emb, const float* __restrict__ rel,
             const float* __restrict__ W1w, const float* __restrict__ W1b,
             const float* __restrict__ fc1w, const float* __restrict__ fc1b,
             const float* __restrict__ bn0g, const float* __restrict__ bn0b,
             const float* __restrict__ bn0m, const float* __restrict__ bn0v,
             const float* __restrict__ bn1g, const float* __restrict__ bn1b,
             const float* __restrict__ bn1m, const float* __restrict__ bn1v,
             _Float16* __restrict__ flat)
{
    __shared__ float es[EDIM], ec[EDIM], rr[EDIM];
    __shared__ float xx[EDIM + 8];
    __shared__ float kf[OC * FW];
    __shared__ float cb[OC];

    const int b = blockIdx.x;
    const int t = threadIdx.x;

    if (t < EDIM) {
        es[t] = emb[(long)e_s[b] * EDIM + t];
        ec[t] = emb[(long)e_c[b] * EDIM + t];
        rr[t] = rel[(long)q_s[b] * EDIM + t];
    }
    __syncthreads();

    if (t < EDIM) {
        float acc = W1b[t];
        for (int k = 0; k < EDIM; ++k) acc += es[k] * W1w[k * EDIM + t];
        for (int k = 0; k < EDIM; ++k) acc += ec[k] * W1w[(EDIM + k) * EDIM + t];
        const float s0 = bn0g[0] * rsqrtf(bn0v[0] + EPSV);
        xx[t] = (acc - bn0m[0]) * s0 + bn0b[0];
    }
    if (t < OC * FW) {
        float acc = fc1b[t];
        for (int k = 0; k < EDIM; ++k) acc += rr[k] * fc1w[k * (OC * FW) + t];
        const int o = t / FW;
        kf[t] = acc * (bn1g[o] * rsqrtf(bn1v[o] + EPSV));
    }
    if (t < OC) {
        const float s1 = bn1g[t] * rsqrtf(bn1v[t] + EPSV);
        cb[t] = bn1b[t] - bn1m[t] * s1;
    }
    __syncthreads();

    for (int i = t; i < FCIN; i += 288) {
        const int o = i / WP, p = i % WP;
        float acc = cb[o];
#pragma unroll
        for (int w = 0; w < FW; ++w) acc += xx[p + w] * kf[o * FW + w];
        flat[(long)b * FCIN + i] = (_Float16)acc;
    }
}

// ---------------------------------------------------------------------------
// Pack fc_w [6144,200] f32 into WMMA B-operand layout, f16:
// pB[ntile][ks][lane][16] where lane L owns column n = ntile*16 + L%16 and
// K = ks*32 + (L<16 ? 0..15 : 16..31) contiguously.
// ---------------------------------------------------------------------------
__global__ __launch_bounds__(256)
void k_pack_fcw(const float* __restrict__ fcw, _Float16* __restrict__ pB)
{
    const long total = (long)NT_FC * KS_FC * 512;
    long idx = (long)blockIdx.x * 256 + threadIdx.x;
    if (idx >= total) return;
    const int j     = (int)(idx & 15);
    const int lane  = (int)((idx >> 4) & 31);
    const long rest = idx >> 9;
    const int ks    = (int)(rest % KS_FC);
    const int nt    = (int)(rest / KS_FC);
    const int n = nt * 16 + (lane & 15);
    const int k = ks * 32 + ((lane < 16) ? 0 : 16) + j;
    const float v = (n < EDIM) ? fcw[(long)k * EDIM + n] : 0.0f;
    pB[idx] = (_Float16)v;
}

// ---------------------------------------------------------------------------
// entity_emb [NE,200] f32 -> f16 padded [NE,224] (zeros for k>=200).
// ---------------------------------------------------------------------------
__global__ __launch_bounds__(256)
void k_pack_emb(const float* __restrict__ emb, _Float16* __restrict__ embh)
{
    const long total = (long)NENT * KPAD;
    long idx = (long)blockIdx.x * 256 + threadIdx.x;
    if (idx >= total) return;
    const long e = idx / KPAD;
    const int  k = (int)(idx % KPAD);
    const float v = (k < EDIM) ? emb[e * EDIM + k] : 0.0f;
    embh[idx] = (_Float16)v;
}

// ---------------------------------------------------------------------------
// h = flat[1024,6144] @ fc_w[6144,200] (+fc_b), bn2, relu -> h f16 [1024,224].
// One wave per 16x16 tile; 64*13 = 832 tiles = 104 blocks * 8 waves.
// ---------------------------------------------------------------------------
__global__ __launch_bounds__(256)
void k_fc_gemm(const _Float16* __restrict__ flat, const _Float16* __restrict__ pB,
               const float* __restrict__ fcb,
               const float* __restrict__ bn2g, const float* __restrict__ bn2b,
               const float* __restrict__ bn2m, const float* __restrict__ bn2v,
               _Float16* __restrict__ hW)
{
    const int tid  = threadIdx.x;
    const int wave = tid >> 5;
    const int lane = tid & 31;
    const int tile = blockIdx.x * 8 + wave;        // 0..831
    const int mtile = tile / NT_FC;
    const int ntile = tile % NT_FC;

    const int mrow = mtile * 16 + (lane & 15);
    const _Float16* Ab = flat + (long)mrow * FCIN + ((lane < 16) ? 0 : 16);
    const _Float16* Bb = pB + ((long)ntile * KS_FC) * 512 + lane * 16;

    v8f c = {};
    for (int ks = 0; ks < KS_FC; ++ks) {
        const int k0 = ks * 32;
        v8h a0 = *(const v8h*)(Ab + k0);
        v8h a1 = *(const v8h*)(Ab + k0 + 8);
        v16h A = __builtin_shufflevector(a0, a1, 0,1,2,3,4,5,6,7,8,9,10,11,12,13,14,15);
        v16h B = *(const v16h*)(Bb + (long)ks * 512);
        c = __builtin_amdgcn_wmma_f32_16x16x32_f16(false, A, false, B,
                                                   (short)0, c, false, false);
    }

    const int nloc = lane & 15;
    const int n = ntile * 16 + nloc;
    const int mbase = mtile * 16 + ((lane < 16) ? 0 : 8);
    if (n < EDIM) {
        const float s2 = bn2g[n] * rsqrtf(bn2v[n] + EPSV);
        const float sh = bn2b[n] - bn2m[n] * s2;
        const float bias = fcb[n];
#pragma unroll
        for (int r = 0; r < 8; ++r) {
            float t = (c[r] + bias) * s2 + sh;
            t = fmaxf(t, 0.0f);
            hW[(long)(mbase + r) * KPAD + n] = (_Float16)t;
        }
    }
}

// ---------------------------------------------------------------------------
// logits = h[1024,224] @ embh^T (+bias), sigmoid -> out f32 [1024,50000].
// Block = 8 waves, each wave owns one M-tile with full-K A in registers;
// B tiles (16 emb rows) staged in LDS and shared by all 8 waves.
// ---------------------------------------------------------------------------
#define NTB 8          // ntiles per block
#define BROW 232       // LDS row stride (f16) -> 464 B, conflict-free

__global__ __launch_bounds__(256)
void k_logits(const _Float16* __restrict__ hW, const _Float16* __restrict__ embh,
              const float* __restrict__ bias_b, float* __restrict__ out)
{
    __shared__ __align__(16) _Float16 Bs[16 * BROW];

    const int ngroups = (NT_LG + NTB - 1) / NTB;   // 391
    const int bx = blockIdx.x;
    const int ng = bx % ngroups;
    const int mg = bx / ngroups;
    const int nt0 = ng * NTB;
    const int ntn = (NT_LG - nt0 < NTB) ? (NT_LG - nt0) : NTB;

    const int tid  = threadIdx.x;
    const int wave = tid >> 5;
    const int lane = tid & 31;
    const int mtile = mg * 8 + wave;               // 0..63

    // Load full-K A operand for this wave's M-tile into registers (7 x v16h).
    const int mrow = mtile * 16 + (lane & 15);
    const _Float16* Ab = hW + (long)mrow * KPAD + ((lane < 16) ? 0 : 8);
    v16h A[KS_LG];
#pragma unroll
    for (int ks = 0; ks < KS_LG; ++ks) {
        v8h a0 = *(const v8h*)(Ab + ks * 32);
        v8h a1 = *(const v8h*)(Ab + ks * 32 + 16);
        A[ks] = __builtin_shufflevector(a0, a1, 0,1,2,3,4,5,6,7,8,9,10,11,12,13,14,15);
    }

    const int nloc = lane & 15;
    const int mbase = mtile * 16 + ((lane < 16) ? 0 : 8);
    const _Float16* Brow = Bs + nloc * BROW + ((lane < 16) ? 0 : 8);

    for (int it = 0; it < ntn; ++it) {
        const int ntile = nt0 + it;
        __syncthreads();   // previous iteration's LDS reads done
        // Cooperative B-tile load: 16 rows x 224 f16 = 448 x 16B chunks.
        for (int i = tid; i < 448; i += 256) {
            const int nl = i / 28, ch = i % 28;
            const uint4* src = (const uint4*)(embh + (long)(ntile * 16 + nl) * KPAD) + ch;
            *((uint4*)(Bs + nl * BROW) + ch) = *src;
        }
        __syncthreads();

        v8f c = {};
#pragma unroll
        for (int ks = 0; ks < KS_LG; ++ks) {
            v8h b0 = *(const v8h*)(Brow + ks * 32);
            v8h b1 = *(const v8h*)(Brow + ks * 32 + 16);
            v16h B = __builtin_shufflevector(b0, b1, 0,1,2,3,4,5,6,7,8,9,10,11,12,13,14,15);
            c = __builtin_amdgcn_wmma_f32_16x16x32_f16(false, A[ks], false, B,
                                                       (short)0, c, false, false);
        }

        const int n = ntile * 16 + nloc;           // < 50000 always
        const float bias = bias_b[n];
#pragma unroll
        for (int r = 0; r < 8; ++r) {
            float t = c[r] + bias;
            t = 1.0f / (1.0f + __expf(-t));
            out[(long)(mbase + r) * NENT + n] = t;
        }
    }
}

// ---------------------------------------------------------------------------
extern "C" void kernel_launch(void* const* d_in, const int* in_sizes, int n_in,
                              void* d_out, int out_size, void* d_ws, size_t ws_size,
                              hipStream_t stream)
{
    const int*   e_s  = (const int*)d_in[0];
    const int*   q_s  = (const int*)d_in[1];
    const int*   e_c  = (const int*)d_in[2];
    const float* emb  = (const float*)d_in[3];
    const float* rel  = (const float*)d_in[4];
    const float* W1w  = (const float*)d_in[5];
    const float* W1b  = (const float*)d_in[6];
    const float* fc1w = (const float*)d_in[7];
    const float* fc1b = (const float*)d_in[8];
    const float* fcw  = (const float*)d_in[9];
    const float* fcb  = (const float*)d_in[10];
    const float* bn0g = (const float*)d_in[11];
    const float* bn0b = (const float*)d_in[12];
    const float* bn0m = (const float*)d_in[13];
    const float* bn0v = (const float*)d_in[14];
    const float* bn1g = (const float*)d_in[15];
    const float* bn1b = (const float*)d_in[16];
    const float* bn1m = (const float*)d_in[17];
    const float* bn1v = (const float*)d_in[18];
    const float* bn2g = (const float*)d_in[19];
    const float* bn2b = (const float*)d_in[20];
    const float* bn2m = (const float*)d_in[21];
    const float* bn2v = (const float*)d_in[22];
    const float* bias_b = (const float*)d_in[23];
    float* out = (float*)d_out;

    // Workspace layout (all offsets 256B-aligned)
    char* ws = (char*)d_ws;
    _Float16* flat = (_Float16*)(ws);                              // 12,582,912 B
    _Float16* pB   = (_Float16*)(ws + 12582912);                   //  2,555,904 B
    _Float16* embh = (_Float16*)(ws + 12582912 + 2555904);         // 22,400,000 B
    _Float16* hW   = (_Float16*)(ws + 12582912 + 2555904 + 22400000); // 458,752 B

    k_front<<<BATCH, 288, 0, stream>>>(e_s, q_s, e_c, emb, rel, W1w, W1b,
                                       fc1w, fc1b, bn0g, bn0b, bn0m, bn0v,
                                       bn1g, bn1b, bn1m, bn1v, flat);

    k_pack_fcw<<<(NT_FC * KS_FC * 512) / 256, 256, 0, stream>>>(fcw, pB);

    k_pack_emb<<<((long)NENT * KPAD) / 256, 256, 0, stream>>>(emb, embh);

    hipMemsetAsync(hW, 0, (size_t)BATCH * KPAD * sizeof(_Float16), stream);

    k_fc_gemm<<<(MT * NT_FC) / 8, 256, 0, stream>>>(flat, pB, fcb,
                                                    bn2g, bn2b, bn2m, bn2v, hW);

    const int ngroups = (NT_LG + NTB - 1) / NTB;   // 391
    k_logits<<<ngroups * 8, 256, 0, stream>>>(hW, embh, bias_b, out);
}
